// Transducer_45535243272325
// MI455X (gfx1250) — compile-verified, matrix operand order checked by reading, not run on hardware
//
#include <hip/hip_runtime.h>
#include <stdint.h>

// ---------------------------------------------------------------------------
// Transducer alignment loss for MI455X (gfx1250, wave32).
//   inputs : [B=64, T=2000, C=512] f32 logits
//   targets: [B=64, L=200] i32
//   out    : scalar f32 loss
// ---------------------------------------------------------------------------

typedef __attribute__((ext_vector_type(16))) _Float16 v16h;
typedef __attribute__((ext_vector_type(8)))  float    v8f;

#define NEGINF (-1.0e30f)

constexpr int Bn = 64;
constexpr int Tn = 2000;
constexpr int Cn = 512;
constexpr int Ln = 200;

// fast logaddexp: max + log1p(exp(min-max)); safe when both args ~ -1e30
__device__ __forceinline__ float lae(float a, float b) {
    float m = fmaxf(a, b);
    float d = fminf(a, b) - m;          // <= 0, finite
    return m + __logf(1.0f + __expf(d));
}

// CDNA5 async global->LDS copy (ASYNCcnt path). VDST = LDS byte address,
// VADDR = 64-bit global address, GV mode (saddr = off).
__device__ __forceinline__ void async_ld_b32(uint32_t lds_off, const void* gaddr) {
    asm volatile("global_load_async_to_lds_b32 %0, %1, off"
                 :: "v"(lds_off), "v"((unsigned long long)(uintptr_t)gaddr)
                 : "memory");
}
#define WAIT_ASYNCCNT(n) asm volatile("s_wait_asynccnt " #n ::: "memory")

// ---------------------------------------------------------------------------
// Kernel 1: per-wave tile of 16 rows of the flattened [B*T, C] logits.
// Phase A: lse via exp -> f16 -> WMMA with all-ones B (row-sum on matrix pipe).
// Phase B: gather the L=200 target logits per row, write emit = x[tgt] - lse.
// ---------------------------------------------------------------------------
__global__ __launch_bounds__(256) void lse_emit_kernel(
        const float* __restrict__ x, const int* __restrict__ tg,
        float* __restrict__ emit) {
    const int lane = threadIdx.x & 31;
    const int wave = threadIdx.x >> 5;
    const int tile = blockIdx.x * 8 + wave;   // 8000 tiles x 16 rows
    const int row0 = tile * 16;
    const int b    = row0 / Tn;               // 2000 % 16 == 0: no tile crosses b
    const int half = lane >> 4;               // lanes 0-15 / 16-31
    const int mrow = lane & 15;               // row this lane feeds into A

    const float* rowp = x + (size_t)(row0 + mrow) * Cn;

    v16h onesv;
#pragma unroll
    for (int i = 0; i < 16; ++i) onesv[i] = (_Float16)1.0f;

    // A-layout (16-bit, 16x32): lanes 0-15 hold K = 0..7 and 16..23,
    // lanes 16-31 hold K = 8..15 and 24..31 -> two 8-float runs per lane.
    v8f acc = {};
#pragma unroll
    for (int k = 0; k < 16; ++k) {
        const int c0 = k * 32 + half * 8;
        float4 f0 = *(const float4*)(rowp + c0);
        float4 f1 = *(const float4*)(rowp + c0 + 4);
        float4 f2 = *(const float4*)(rowp + c0 + 16);
        float4 f3 = *(const float4*)(rowp + c0 + 20);
        v16h a;
        a[0]  = (_Float16)__expf(f0.x);  a[1]  = (_Float16)__expf(f0.y);
        a[2]  = (_Float16)__expf(f0.z);  a[3]  = (_Float16)__expf(f0.w);
        a[4]  = (_Float16)__expf(f1.x);  a[5]  = (_Float16)__expf(f1.y);
        a[6]  = (_Float16)__expf(f1.z);  a[7]  = (_Float16)__expf(f1.w);
        a[8]  = (_Float16)__expf(f2.x);  a[9]  = (_Float16)__expf(f2.y);
        a[10] = (_Float16)__expf(f2.z);  a[11] = (_Float16)__expf(f2.w);
        a[12] = (_Float16)__expf(f3.x);  a[13] = (_Float16)__expf(f3.y);
        a[14] = (_Float16)__expf(f3.z);  a[15] = (_Float16)__expf(f3.w);
        acc = __builtin_amdgcn_wmma_f32_16x16x32_f16(
                  false, a, false, onesv, (short)0, acc, false, false);
    }

    // D layout: lane L (half h = L>>4) holds rows M = r + 8h in acc[r].
    // Lane L wants the sum of row (L & 15): pick acc[L&7] on lane
    // src = (L&7) + 16*((L>>3)&1), whose row is (L&7) + 8*((L>>3)&1) = L&15.
    const int r = lane & 7;
    float v = acc[0];
    v = (r == 1) ? acc[1] : v;
    v = (r == 2) ? acc[2] : v;
    v = (r == 3) ? acc[3] : v;
    v = (r == 4) ? acc[4] : v;
    v = (r == 5) ? acc[5] : v;
    v = (r == 6) ? acc[6] : v;
    v = (r == 7) ? acc[7] : v;
    const int srcl = r + (((lane >> 3) & 1) << 4);
    float rowsum = __shfl(v, srcl, 32);
    float lse = __logf(rowsum);               // lane L: lse of row (L & 15)

    // Phase B: gather 200 target columns per row; rows are hot in cache.
    const int* tb = tg + b * Ln;
    for (int l0 = 0; l0 < Ln; l0 += 32) {
        const int  l   = l0 + lane;
        const bool act = (l < Ln);
        const int  t   = act ? tb[l] : 0;
#pragma unroll 4
        for (int m = 0; m < 16; ++m) {
            float lsem = __shfl(lse, m, 32);  // row m's lse
            float val  = x[(size_t)(row0 + m) * Cn + t];
            if (act) emit[(size_t)(row0 + m) * Ln + l] = val - lsem;
        }
    }
}

// ---------------------------------------------------------------------------
// Kernel 2: DP over t. One block per batch; thread l owns alpha[l].
// Emit rows prefetched 8 deep via async global->LDS (ASYNCcnt) ring buffer.
// alpha[l-1]: shfl_up in-wave, double-buffered LDS slot across wave edges.
// ---------------------------------------------------------------------------
#define AD 8                                  // async ring depth (power of 2)

__global__ __launch_bounds__(256) void dp_kernel(
        const float* __restrict__ emit, float* __restrict__ out) {
    __shared__ float ring[AD][256];
    __shared__ float bnd[2][8];

    const int tid  = threadIdx.x;
    const int lane = tid & 31;
    const int wave = tid >> 5;
    const int b    = blockIdx.x;
    const int col  = (tid < Ln) ? tid : (Ln - 1);   // clamp dead lanes
    const float* eb = emit + (size_t)b * Tn * Ln;

    // prologue: fill the ring for t = 0..AD-1
#pragma unroll
    for (int t = 0; t < AD; ++t)
        async_ld_b32((uint32_t)(uintptr_t)&ring[t][tid],
                     eb + (size_t)t * Ln + col);

    // t = 0 : alpha = emit0 + (l==0 ? 0 : -inf)
    WAIT_ASYNCCNT(7);
    float e0    = (tid < Ln) ? ring[0][tid] : NEGINF;
    float alpha = e0 + ((tid == 0) ? 0.0f : NEGINF);
    if (lane == 31) bnd[0][wave] = alpha;
    __syncthreads();

    for (int t = 1; t < Tn; ++t) {
        // keep exactly AD loads in flight; past the end, re-issue a clamped
        // dummy into an already-consumed slot so the constant wait stays valid
        const int j  = t + AD - 1;
        const int jr = (j < Tn) ? j : (Tn - 1);
        async_ld_b32((uint32_t)(uintptr_t)&ring[j & (AD - 1)][tid],
                     eb + (size_t)jr * Ln + col);
        WAIT_ASYNCCNT(7);                     // in-order: slot t&(AD-1) ready

        float e  = (tid < Ln) ? ring[t & (AD - 1)][tid] : NEGINF;
        float up = __shfl_up(alpha, 1, 32);
        if (lane == 0) up = (wave == 0) ? NEGINF : bnd[(t - 1) & 1][wave - 1];
        alpha = e + lae(alpha, up);
        if (lane == 31) bnd[t & 1][wave] = alpha;
        __syncthreads();
    }

    if (tid == Ln - 1) atomicAdd(out, -alpha * (1.0f / (float)Bn));
}

__global__ void zero_out_kernel(float* o) {
    if (threadIdx.x == 0) o[0] = 0.0f;
}

// ---------------------------------------------------------------------------
extern "C" void kernel_launch(void* const* d_in, const int* in_sizes, int n_in,
                              void* d_out, int out_size, void* d_ws, size_t ws_size,
                              hipStream_t stream) {
    const float* x  = (const float*)d_in[0];   // [B,T,C] f32
    const int*   tg = (const int*)d_in[1];     // [B,L]   i32
    float* out  = (float*)d_out;
    float* emit = (float*)d_ws;                // [B*T, L] f32 = 102.4 MB

    zero_out_kernel<<<1, 32, 0, stream>>>(out);
    // 128000 rows / 16 per wave = 8000 waves; 8 waves per 256-thread block
    lse_emit_kernel<<<1000, 256, 0, stream>>>(x, tg, emit);
    dp_kernel<<<Bn, 256, 0, stream>>>(emit, out);
}
